// AttentionAugmentation2D_27187142983955
// MI455X (gfx1250) — matrix operand order repaired and checked by reading, not compiled, not run on hardware
//
#include <hip/hip_runtime.h>

// ---------- types ----------
typedef __attribute__((ext_vector_type(16))) __bf16 v16bf;
typedef __attribute__((ext_vector_type(8)))  float  v8f;

union BF16x16 {
    v16bf        bf;
    unsigned int u[8];
    uint4        u4[2];
};

__device__ __forceinline__ unsigned short f32_to_bf16(float f) {
    unsigned int u = __float_as_uint(f);
    u = (u + 0x7FFFu + ((u >> 16) & 1u)) >> 16;   // round-to-nearest-even
    return (unsigned short)u;
}
__device__ __forceinline__ float bf16_to_f32(unsigned short h) {
    return __uint_as_float(((unsigned int)h) << 16);
}

// Problem constants: B=8, Nh=8, H=W=32, L=1024, d=32
#define NH    8
#define LTOK  1024
#define DHEAD 32

// ---------- kernel 1: split heads, scale q, bf16-convert, pre-swizzle K ----------
// Kt is stored directly in WMMA B-fragment order:
//   Kt[((bn*64 + frag)*32 + ((d>>3)&1)*16 + h)*16 + (d&7) + (d>>4)*8] = K[key = frag*16 + h][d]
__global__ __launch_bounds__(256) void conv_kernel(
    const float* __restrict__ in,        // [B,32,32,768]
    unsigned short* __restrict__ Q,      // [B,Nh,L,32] bf16 (scaled)
    unsigned short* __restrict__ Kt,     // [B*Nh,64,32,16] bf16 (fragment order)
    unsigned short* __restrict__ V)      // [B*Nh,32,32,32] bf16 (fragment order)
{
    int t   = blockIdx.x * 256 + threadIdx.x;      // 2,097,152 threads
    int d   = t & 31;
    int pos = (t >> 5) & 1023;
    int bn  = t >> 15;                             // (b*8+n)
    size_t ibase = ((size_t)((bn >> 3) * 1024 + pos)) * 768 + (bn & 7) * 32 + d;
    const float scale = 0.17677669529663687f;      // (256/8)^-0.5
    Q[t] = f32_to_bf16(in[ibase] * scale);
    { int kk = pos & 63, kw = kk & 31; V[((size_t)bn * 1024 + (pos >> 6) * 64 + (kk >> 5) * 32 + ((kw >> 3) & 1) * 16 + (d & 15)) * 32 + (d >> 4) * 16 + (kw & 7) + ((kw >> 4) << 3)] = f32_to_bf16(in[ibase + 512]); }
    int frag = pos >> 4, h = pos & 15;
    Kt[(((size_t)bn * 64 + frag) * 32 + ((d >> 3) & 1) * 16 + h) * 16 + (d & 7) + ((d >> 4) << 3)] = f32_to_bf16(in[ibase + 256]);
}

// ---------- kernel 2: separable relative-position bias table ----------
// RB[bn][q][j], j<32: q.rel_h[j - x + 31] ; j>=32: q.rel_w[(j-32) - w + 31]
__global__ __launch_bounds__(256) void bias_kernel(
    const unsigned short* __restrict__ Q,
    const float* __restrict__ relw,      // [63,32]
    const float* __restrict__ relh,      // [63,32]
    float* __restrict__ RB)              // [64,1024,64]
{
    int t  = blockIdx.x * 256 + threadIdx.x;       // 4,194,304 threads
    int j  = t & 63;
    int q  = (t >> 6) & 1023;
    int bn = t >> 16;
    int x = q >> 5, w = q & 31;
    const float* rel = (j < 32) ? (relh + (j - x + 31) * 32)
                                : (relw + ((j - 32) - w + 31) * 32);
    const unsigned short* qr = Q + ((size_t)bn * 1024 + q) * 32;
    float s = 0.f;
#pragma unroll
    for (int d = 0; d < 32; ++d)
        s += bf16_to_f32(qr[d]) * rel[d];
    RB[t] = s;
}

// ---------- kernel 3: flash attention with WMMA ----------
// one wave32 == one 16-query tile; 64-key chunks; 8 waves / block; 512 blocks.
__global__ __launch_bounds__(256) void flash_kernel(
    const unsigned short* __restrict__ Q,
    const unsigned short* __restrict__ Kt,
    const unsigned short* __restrict__ V,
    const float* __restrict__ RB,
    float* __restrict__ out)             // [B,32,32,256] f32
{
    __shared__ float          s_rb[8][16 * 64];      // bias tile per wave
    __shared__ unsigned short s_p [8][16 * 64];      // P staging per wave

    const int lane = threadIdx.x & 31;
    const int wid  = threadIdx.x >> 5;
    const int gw   = blockIdx.x * 8 + wid;           // 0..4095
    const int qt   = gw & 63;
    const int bn   = gw >> 6;                        // b*8+n
    const int q0   = qt * 16;

    const int g  = lane >> 4;                        // lane half-group
    const int ln = lane & 15;

    const size_t headoff = (size_t)bn * LTOK * DHEAD;
    const unsigned short* Qh  = Q  + headoff;
    const unsigned short* Vh  = V  + headoff;
    const unsigned short* Kth = Kt + headoff;        // 64*32*16 == 1024*32
    const float* RBh = RB + (size_t)bn * LTOK * 64;

    // bias tile -> LDS (coalesced)
    for (int i = lane; i < 16 * 64; i += 32)
        s_rb[wid][i] = RBh[q0 * 64 + i];
    asm volatile("s_wait_dscnt 0" ::: "memory");

    // Q A-fragment (ISA 16-bit A layout: lanes0-15 K={0..7,16..23}, lanes16-31 K={8..15,24..31})
    BF16x16 aQ;
#pragma unroll
    for (int v = 0; v < 8; ++v) {
        int kb = 2 * v + ((v < 4) ? (g ? 8 : 0) : (g ? 16 : 8));
        aQ.u[v] = *(const unsigned int*)(Qh + (q0 + ln) * DHEAD + kb);
    }

    // all-ones B fragment for matrix-unit row sums
    BF16x16 bOne;
#pragma unroll
    for (int v = 0; v < 8; ++v) bOne.u[v] = 0x3F803F80u;   // bf16 1.0 pair

    v8f O0 = {}, O1 = {};
    float m8[8], l8[8], rw0[8], rw1[8];
#pragma unroll
    for (int r = 0; r < 8; ++r) {
        m8[r] = -3.0e38f; l8[r] = 0.f;
        int M = r + 8 * g;
        rw0[r] = s_rb[wid][M * 64 + 32 + ln];        // rel_w bias, cols 0..15
        rw1[r] = s_rb[wid][M * 64 + 48 + ln];        // rel_w bias, cols 16..31
    }

    for (int j = 0; j < 16; ++j) {                   // 64 keys: key-rows 2j, 2j+1
        const int kc = j * 64;
        const unsigned short* kp = Kth + ((size_t)(4 * j) * 32 + lane) * 16;
        const unsigned short* vp = Vh + (size_t)(kc + lane) * DHEAD;

        if (j < 15) {                                // gfx1250 global_prefetch
            __builtin_prefetch(kp + 2048, 0, 1);     // next chunk's K fragments
            __builtin_prefetch(vp + 2048, 0, 1);     // next chunk's V rows
        }

        // K^T B-fragments (pre-swizzled): 2 x b128 per lane per fragment
        BF16x16 bK0, bK1, bK2, bK3;
        bK0.u4[0] = *(const uint4*)(kp);        bK0.u4[1] = *(const uint4*)(kp + 8);
        bK1.u4[0] = *(const uint4*)(kp + 512);  bK1.u4[1] = *(const uint4*)(kp + 520);
        bK2.u4[0] = *(const uint4*)(kp + 1024); bK2.u4[1] = *(const uint4*)(kp + 1032);
        bK3.u4[0] = *(const uint4*)(kp + 1536); bK3.u4[1] = *(const uint4*)(kp + 1544);

        // bias rides the WMMA accumulator: C[r] = rel_h(key-row) + rel_w(col)
        v8f C0, C1, C2, C3;
#pragma unroll
        for (int r = 0; r < 8; ++r) {
            int M = r + 8 * g;
            float rha = s_rb[wid][M * 64 + 2 * j];
            float rhb = s_rb[wid][M * 64 + 2 * j + 1];
            C0[r] = rha + rw0[r];
            C1[r] = rha + rw1[r];
            C2[r] = rhb + rw0[r];
            C3[r] = rhb + rw1[r];
        }
        v8f S0 = __builtin_amdgcn_wmma_f32_16x16x32_bf16(false, aQ.bf, false, bK0.bf, (short)0, C0, false, false);
        v8f S1 = __builtin_amdgcn_wmma_f32_16x16x32_bf16(false, aQ.bf, false, bK1.bf, (short)0, C1, false, false);
        v8f S2 = __builtin_amdgcn_wmma_f32_16x16x32_bf16(false, aQ.bf, false, bK2.bf, (short)0, C2, false, false);
        v8f S3 = __builtin_amdgcn_wmma_f32_16x16x32_bf16(false, aQ.bf, false, bK3.bf, (short)0, C3, false, false);

        // online softmax: per-row max via 16-lane shuffle tree; exp; stage P
        v8f lC;
#pragma unroll
        for (int r = 0; r < 8; ++r) {
            int M = r + 8 * g;
            float rm = fmaxf(fmaxf(S0[r], S1[r]), fmaxf(S2[r], S3[r]));
#pragma unroll
            for (int mm = 1; mm < 16; mm <<= 1)
                rm = fmaxf(rm, __shfl_xor(rm, mm, 32));
            float mn = fmaxf(m8[r], rm);
            float alpha = __expf(m8[r] - mn);
            m8[r] = mn;
            float p0 = __expf(S0[r] - mn);
            float p1 = __expf(S1[r] - mn);
            float p2 = __expf(S2[r] - mn);
            float p3 = __expf(S3[r] - mn);
            lC[r] = l8[r] * alpha;
            O0[r] *= alpha;
            O1[r] *= alpha;
            s_p[wid][M * 64 + ln]      = f32_to_bf16(p0);   // C-layout -> LDS
            s_p[wid][M * 64 + 16 + ln] = f32_to_bf16(p1);
            s_p[wid][M * 64 + 32 + ln] = f32_to_bf16(p2);
            s_p[wid][M * 64 + 48 + ln] = f32_to_bf16(p3);
        }
        asm volatile("s_wait_dscnt 0" ::: "memory");        // in-wave DS ordering

        // P A-fragments from LDS (transpose to A layout), keys 0..31 / 32..63
        BF16x16 aP0, aP1;
#pragma unroll
        for (int v = 0; v < 8; ++v) {
            int kb = 2 * v + ((v < 4) ? (g ? 8 : 0) : (g ? 16 : 8));
            aP0.u[v] = *(const unsigned int*)&s_p[wid][ln * 64 + kb];
            aP1.u[v] = *(const unsigned int*)&s_p[wid][ln * 64 + 32 + kb];
        }

        // V B-fragments (pre-swizzled): lane%16 = d, lane half + halfword = key (table order)
        BF16x16 bVa0, bVa1, bVb0, bVb1;
        bVa0.u4[0] = *(const uint4*)(vp);         bVa0.u4[1] = *(const uint4*)(vp + 8);
        bVa1.u4[0] = *(const uint4*)(vp + 16);    bVa1.u4[1] = *(const uint4*)(vp + 24);
        bVb0.u4[0] = *(const uint4*)(vp + 1024);  bVb0.u4[1] = *(const uint4*)(vp + 1032);
        bVb1.u4[0] = *(const uint4*)(vp + 1040);  bVb1.u4[1] = *(const uint4*)(vp + 1048);

        O0 = __builtin_amdgcn_wmma_f32_16x16x32_bf16(false, aP0.bf, false, bVa0.bf, (short)0, O0, false, false);
        O1 = __builtin_amdgcn_wmma_f32_16x16x32_bf16(false, aP0.bf, false, bVa1.bf, (short)0, O1, false, false);
        O0 = __builtin_amdgcn_wmma_f32_16x16x32_bf16(false, aP1.bf, false, bVb0.bf, (short)0, O0, false, false);
        O1 = __builtin_amdgcn_wmma_f32_16x16x32_bf16(false, aP1.bf, false, bVb1.bf, (short)0, O1, false, false);

        // l update on the matrix unit: l_new = l_old*alpha + P x ones
        v8f lw = __builtin_amdgcn_wmma_f32_16x16x32_bf16(false, aP0.bf, false, bOne.bf, (short)0, lC, false, false);
        lw     = __builtin_amdgcn_wmma_f32_16x16x32_bf16(false, aP1.bf, false, bOne.bf, (short)0, lw, false, false);
#pragma unroll
        for (int r = 0; r < 8; ++r) l8[r] = lw[r];
    }

    // normalize + write out[b, x, w, n*32 + d]
    const int b = bn >> 3, n = bn & 7;
#pragma unroll
    for (int r = 0; r < 8; ++r) {
        int M = r + 8 * g;
        int q = q0 + M;
        int x = q >> 5, w = q & 31;
        float inv = 1.0f / l8[r];
        size_t o = (((size_t)(b * 32 + x)) * 32 + w) * 256 + n * 32;
        out[o + ln]      = O0[r] * inv;
        out[o + 16 + ln] = O1[r] * inv;
    }
}

// ---------- launcher ----------
extern "C" void kernel_launch(void* const* d_in, const int* in_sizes, int n_in,
                              void* d_out, int out_size, void* d_ws, size_t ws_size,
                              hipStream_t stream) {
    const float* in   = (const float*)d_in[0];   // [8,32,32,768]
    const float* relw = (const float*)d_in[1];   // [63,32]
    const float* relh = (const float*)d_in[2];   // [63,32]
    float* out = (float*)d_out;                  // [8,32,32,256]

    const size_t NE = (size_t)8 * NH * LTOK * DHEAD;   // 2,097,152
    unsigned short* Q  = (unsigned short*)d_ws;        // 4 MB
    unsigned short* Kt = Q + NE;                       // 4 MB (fragment order)
    unsigned short* Vp = Kt + NE;                      // 4 MB
    float* RB = (float*)(Vp + NE);                     // 16 MB

    conv_kernel <<<8192, 256, 0, stream>>>(in, Q, Kt, Vp);
    bias_kernel <<<16384, 256, 0, stream>>>(Q, relw, relh, RB);
    flash_kernel<<<512, 256, 0, stream>>>(Q, Kt, Vp, RB, out);
}
